// GAT_8177617731651
// MI455X (gfx1250) — matrix operand order repaired and checked by reading, not compiled, not run on hardware
//
#include <hip/hip_runtime.h>
#include <math.h>

#define NEG_SLOPE 0.2f

typedef __bf16 bf16_t;
typedef bf16_t v16bf __attribute__((ext_vector_type(16)));
typedef float  v8f   __attribute__((ext_vector_type(8)));

__device__ __forceinline__ unsigned short f2bf(float f) {
  unsigned int u = __float_as_uint(f);
  unsigned int r = 0x7FFFu + ((u >> 16) & 1u);   // round-to-nearest-even
  return (unsigned short)((u + r) >> 16);
}

__device__ __forceinline__ void atomicMaxF(float* addr, float val) {
  if (val >= 0.0f) atomicMax((int*)addr, __float_as_int(val));
  else             atomicMin((unsigned int*)addr, __float_as_uint(val));
}

// ---------------- utility kernels ----------------

__global__ void fill_f32(float* __restrict__ p, float v, long n) {
  long i = (long)blockIdx.x * blockDim.x + threadIdx.x;
  if (i < n) p[i] = v;
}

// Pack B[K,NCOL] fp32 -> bf16 in per-lane WMMA fragment order:
// out[(((kb/32)*(NCOL/16) + nt)*32 + lane)*16 + s] = bf16(B[kb + (lane>>4)*16 + s][nt*16 + (lane&15)])
__global__ void pack_b_bf16_k(const float* __restrict__ W, unsigned short* __restrict__ out,
                              int K, int NCOL) {
  int i = blockIdx.x * blockDim.x + threadIdx.x;
  if (i >= K * NCOL) return;
  const int ntiles = NCOL >> 4;
  int s    = i & 15;
  int lane = (i >> 4) & 31;
  int tile = i >> 9;
  int kb32 = tile / ntiles;
  int nt   = tile - kb32 * ntiles;
  int k = (kb32 << 5) + ((lane >> 4) << 4) + s;
  int n = (nt << 4) + (lane & 15);
  out[i] = f2bf(W[k * NCOL + n]);
}

// ---------------- WMMA GEMM: C[M,NCOL] = A[M,K] x B[K,NCOL] ----------------
// A: fp32, converted to bf16 while staging in LDS (padded stride, vectorized).
// Bp: pre-swizzled bf16 fragments (see pack_b_bf16_k) -> 2x b128 loads per k-step.
// blockDim.x = 32 * (NCOL/16); one 16-row tile per block; one wave per 16-col tile.
__global__ void gemm_bf16_wmma(const float* __restrict__ A, const unsigned short* __restrict__ Bp,
                               float* __restrict__ C, int M, int K, int NCOL) {
  extern __shared__ unsigned short As[];        // 16 * (K+8) ushorts
  const int KP = K + 8;                         // padded stride (multiple of 8 -> 16B aligned rows)
  const int m0 = blockIdx.x * 16;
  const int tid = threadIdx.x;

  // vectorized cooperative stage of the 16xK A tile as packed bf16
  {
    const int kv = K >> 2;                      // float4s per row
    for (int i = tid; i < 16 * kv; i += blockDim.x) {
      int r = i / kv, k4 = (i - r * kv) << 2;
      int mm = m0 + r;
      float4 f = make_float4(0.f, 0.f, 0.f, 0.f);
      if (mm < M) f = *(const float4*)(A + (long)mm * K + k4);
      uint2 u;
      u.x = (unsigned int)f2bf(f.x) | ((unsigned int)f2bf(f.y) << 16);
      u.y = (unsigned int)f2bf(f.z) | ((unsigned int)f2bf(f.w) << 16);
      *(uint2*)(As + r * KP + k4) = u;
    }
  }
  __syncthreads();

  const int lane   = tid & 31;
  const int wave   = tid >> 5;
  const int ntiles = NCOL >> 4;
  const int n0     = wave * 16;
  const int row    = lane & 15;    // A row / D column
  const int hs     = lane >> 4;    // half-select

  v8f acc = {};
  for (int kb = 0; kb < K; kb += 32) {
    union { v16bf v; uint4 q[2]; } a, b;
    // A fragment: one contiguous 32B run in LDS (k = 16*hs + s)
    const unsigned short* ap = As + row * KP + kb + hs * 16;
    a.q[0] = *(const uint4*)(ap);
    a.q[1] = *(const uint4*)(ap + 8);
    // B fragment: one contiguous 32B run in pre-swizzled global
    const unsigned short* bp = Bp + (((long)(kb >> 5) * ntiles + wave) * 32 + lane) * 16;
    b.q[0] = *(const uint4*)(bp);
    b.q[1] = *(const uint4*)(bp + 8);
    acc = __builtin_amdgcn_wmma_f32_16x16x32_bf16(false, a.v, false, b.v,
                                                  (short)0, acc, false, false);
  }
  // D layout: VGPR j of lane L -> (M = j + 8*(L/16), N = L%16)
#pragma unroll
  for (int j = 0; j < 8; ++j) {
    int mm = m0 + j + 8 * hs;
    if (mm < M) C[(long)mm * NCOL + n0 + row] = acc[j];
  }
}

// ---------------- attention logits: a = sum_c feats[n,h,c] * att[h,c] ----------------
template <int H, int Cc>
__global__ void att_logits_k(const float* __restrict__ feats, const float* __restrict__ att_src,
                             const float* __restrict__ att_dst, float* __restrict__ a_src,
                             float* __restrict__ a_dst, long N) {
  long t = (long)blockIdx.x * blockDim.x + threadIdx.x;
  if (t >= N * H) return;
  long n = t / H; int h = (int)(t - n * H);
  const float* f = feats + (n * H + h) * (long)Cc;
  float s0 = 0.f, s1 = 0.f;
#pragma unroll
  for (int c = 0; c < Cc; ++c) {
    float v = f[c];
    s0 += v * att_src[h * Cc + c];
    s1 += v * att_dst[h * Cc + c];
  }
  a_src[t] = s0;
  a_dst[t] = s1;
}

__device__ __forceinline__ void edge_ids(const long long* __restrict__ ei, long e, long E,
                                         int& s, int& d) {
  if (e < E) { s = (int)ei[e]; d = (int)ei[E + e]; }
  else       { s = d = (int)(e - E); }          // self-loop
}

// ---------------- edge pass 1: segment max ----------------
template <int H>
__global__ void edge_max_k(const long long* __restrict__ ei, long E, long Etot,
                           const float* __restrict__ a_src, const float* __restrict__ a_dst,
                           float* __restrict__ m) {
  long e = (long)blockIdx.x * blockDim.x + threadIdx.x;
  if (e >= Etot) return;
  int s, d; edge_ids(ei, e, E, s, d);
#pragma unroll
  for (int h = 0; h < H; ++h) {
    float x = a_src[(long)s * H + h] + a_dst[(long)d * H + h];
    x = (x > 0.f) ? x : NEG_SLOPE * x;
    atomicMaxF(&m[(long)d * H + h], x);
  }
}

// ---------------- edge pass 2: softmax denominator ----------------
template <int H>
__global__ void edge_den_k(const long long* __restrict__ ei, long E, long Etot,
                           const float* __restrict__ a_src, const float* __restrict__ a_dst,
                           const float* __restrict__ m, float* __restrict__ den) {
  long e = (long)blockIdx.x * blockDim.x + threadIdx.x;
  if (e >= Etot) return;
  int s, d; edge_ids(ei, e, E, s, d);
#pragma unroll
  for (int h = 0; h < H; ++h) {
    float x = a_src[(long)s * H + h] + a_dst[(long)d * H + h];
    x = (x > 0.f) ? x : NEG_SLOPE * x;
    atomicAdd(&den[(long)d * H + h], expf(x - m[(long)d * H + h]));
  }
}

// ---------------- edge pass 3: weighted aggregate (thread per (edge, head)) ----------------
template <int H, int Cc>
__global__ void edge_agg_k(const long long* __restrict__ ei, long E, long Etot,
                           const float* __restrict__ a_src, const float* __restrict__ a_dst,
                           const float* __restrict__ m, const float* __restrict__ den,
                           const float* __restrict__ feats, float* __restrict__ agg) {
  long t = (long)blockIdx.x * blockDim.x + threadIdx.x;
  if (t >= Etot * H) return;
  long e = t / H; int h = (int)(t - e * H);
  int s, d; edge_ids(ei, e, E, s, d);
  float x = a_src[(long)s * H + h] + a_dst[(long)d * H + h];
  x = (x > 0.f) ? x : NEG_SLOPE * x;
  long di = (long)d * H + h;
  float alpha = expf(x - m[di]) / den[di];
  const float* fs = feats + ((long)s * H + h) * (long)Cc;
  float* out = agg + di * (long)Cc;
#pragma unroll
  for (int c = 0; c < Cc; c += 4) {
    float4 f = *(const float4*)(fs + c);
    atomicAdd(&out[c + 0], alpha * f.x);
    atomicAdd(&out[c + 1], alpha * f.y);
    atomicAdd(&out[c + 2], alpha * f.z);
    atomicAdd(&out[c + 3], alpha * f.w);
  }
}

// ---------------- elu(agg + bias) ----------------
template <int F>
__global__ void elu_bias_k(const float* __restrict__ agg, const float* __restrict__ bias,
                           float* __restrict__ out, long N) {
  long i = (long)blockIdx.x * blockDim.x + threadIdx.x;
  if (i >= N * F) return;
  int j = (int)(i & (F - 1));
  float v = agg[i] + bias[j];
  out[i] = (v > 0.f) ? v : (expf(v) - 1.f);
}

// ---------------- final: bias + log_softmax over classes ----------------
template <int C>
__global__ void final_k(const float* __restrict__ agg2, const float* __restrict__ bias2,
                        float* __restrict__ out, long N) {
  long n = (long)blockIdx.x * blockDim.x + threadIdx.x;
  if (n >= N) return;
  float v[C];
  float mx = -INFINITY;
#pragma unroll
  for (int c = 0; c < C; ++c) { v[c] = agg2[n * C + c] + bias2[c]; mx = fmaxf(mx, v[c]); }
  float se = 0.f;
#pragma unroll
  for (int c = 0; c < C; ++c) se += expf(v[c] - mx);
  float lse = mx + logf(se);
#pragma unroll
  for (int c = 0; c < C; ++c) out[n * C + c] = v[c] - lse;
}

// ---------------- host launch ----------------
extern "C" void kernel_launch(void* const* d_in, const int* in_sizes, int n_in,
                              void* d_out, int out_size, void* d_ws, size_t ws_size,
                              hipStream_t stream) {
  const float*     x        = (const float*)d_in[0];
  const long long* ei       = (const long long*)d_in[1];   // int64 [2, E]
  const float*     W1       = (const float*)d_in[2];
  const float*     att_src1 = (const float*)d_in[3];
  const float*     att_dst1 = (const float*)d_in[4];
  const float*     bias1    = (const float*)d_in[5];
  const float*     W2       = (const float*)d_in[6];
  const float*     att_src2 = (const float*)d_in[7];
  const float*     att_dst2 = (const float*)d_in[8];
  const float*     bias2    = (const float*)d_in[9];

  const int  FIN = 512, H1 = 8, C1 = 8, F1 = 64, CLSN = 16;
  const long N    = in_sizes[0] / FIN;
  const long E    = in_sizes[1] / 2;
  const long Etot = E + N;

  // carve workspace (256B aligned chunks)
  char* p = (char*)d_ws;
  auto carve = [&](size_t bytes) { void* r = (void*)p; p += (bytes + 255) & ~(size_t)255; return r; };
  float*          h1     = (float*)carve((size_t)N * F1 * 4);
  float*          asrc1  = (float*)carve((size_t)N * H1 * 4);
  float*          adst1  = (float*)carve((size_t)N * H1 * 4);
  float*          m1     = (float*)carve((size_t)N * H1 * 4);
  float*          den1   = (float*)carve((size_t)N * H1 * 4);
  float*          agg1   = (float*)carve((size_t)N * F1 * 4);
  float*          h2in   = (float*)carve((size_t)N * F1 * 4);
  float*          feat2  = (float*)carve((size_t)N * CLSN * 4);
  float*          asrc2  = (float*)carve((size_t)N * 4);
  float*          adst2  = (float*)carve((size_t)N * 4);
  float*          m2     = (float*)carve((size_t)N * 4);
  float*          den2   = (float*)carve((size_t)N * 4);
  float*          agg2   = (float*)carve((size_t)N * CLSN * 4);
  unsigned short* W1p    = (unsigned short*)carve((size_t)FIN * F1 * 2);
  unsigned short* W2p    = (unsigned short*)carve((size_t)F1 * CLSN * 2);
  float*          outp   = (float*)d_out;

  const int BT = 256;
  auto nb = [&](long n) { return (int)((n + BT - 1) / BT); };

  // weight conversion + fragment-order swizzle to bf16
  pack_b_bf16_k<<<nb(FIN * F1), BT, 0, stream>>>(W1, W1p, FIN, F1);
  pack_b_bf16_k<<<nb(F1 * CLSN), BT, 0, stream>>>(W2, W2p, F1, CLSN);

  // initialize accumulators
  fill_f32<<<nb(N * H1), BT, 0, stream>>>(m1, -INFINITY, N * H1);
  fill_f32<<<nb(N * H1), BT, 0, stream>>>(den1, 0.f, N * H1);
  fill_f32<<<nb(N * F1), BT, 0, stream>>>(agg1, 0.f, N * F1);
  fill_f32<<<nb(N), BT, 0, stream>>>(m2, -INFINITY, N);
  fill_f32<<<nb(N), BT, 0, stream>>>(den2, 0.f, N);
  fill_f32<<<nb(N * CLSN), BT, 0, stream>>>(agg2, 0.f, N * CLSN);

  // ---- layer 1 ----
  {
    int mtiles = (int)((N + 15) / 16);
    int blk = 32 * (F1 / 16);                       // 128 threads = 4 waves
    size_t shmem = (size_t)16 * (FIN + 8) * 2;
    gemm_bf16_wmma<<<mtiles, blk, shmem, stream>>>(x, W1p, h1, (int)N, FIN, F1);
  }
  att_logits_k<8, 8><<<nb(N * H1), BT, 0, stream>>>(h1, att_src1, att_dst1, asrc1, adst1, N);
  edge_max_k<8><<<nb(Etot), BT, 0, stream>>>(ei, E, Etot, asrc1, adst1, m1);
  edge_den_k<8><<<nb(Etot), BT, 0, stream>>>(ei, E, Etot, asrc1, adst1, m1, den1);
  edge_agg_k<8, 8><<<nb(Etot * H1), BT, 0, stream>>>(ei, E, Etot, asrc1, adst1, m1, den1, h1, agg1);
  elu_bias_k<64><<<nb(N * F1), BT, 0, stream>>>(agg1, bias1, h2in, N);

  // ---- layer 2 ----
  {
    int mtiles = (int)((N + 15) / 16);
    int blk = 32 * (CLSN / 16);                     // 32 threads = 1 wave
    size_t shmem = (size_t)16 * (F1 + 8) * 2;
    gemm_bf16_wmma<<<mtiles, blk, shmem, stream>>>(h2in, W2p, feat2, (int)N, F1, CLSN);
  }
  att_logits_k<1, 16><<<nb(N), BT, 0, stream>>>(feat2, att_src2, att_dst2, asrc2, adst2, N);
  edge_max_k<1><<<nb(Etot), BT, 0, stream>>>(ei, E, Etot, asrc2, adst2, m2);
  edge_den_k<1><<<nb(Etot), BT, 0, stream>>>(ei, E, Etot, asrc2, adst2, m2, den2);
  edge_agg_k<1, 16><<<nb(Etot), BT, 0, stream>>>(ei, E, Etot, asrc2, adst2, m2, den2, feat2, agg2);

  final_k<16><<<nb(N), BT, 0, stream>>>(agg2, bias2, outp, N);
}
